// RSSM_20675972563648
// MI455X (gfx1250) — matrix operand order repaired and checked by reading, not compiled, not run on hardware
//
#include <hip/hip_runtime.h>
#include <hip/hip_bf16.h>

// ---------------------------------------------------------------------------
// RSSM step for MI455X (gfx1250): all six GEMMs via v_wmma_f32_16x16x32_bf16.
// Operands kept in WMMA fragment-order so every fragment is a contiguous
// 1KB block, read as 2x ds_load_b128 per lane (quads 512B apart). Block tile 256x128, wave tile 64x64.
// Staging path A (preferred): GLOBAL_LOAD_ASYNC_TO_LDS_B128 (ASYNCcnt) --
// tile data never touches VGPRs. Path B (fallback): named-register pipeline.
// ---------------------------------------------------------------------------

typedef __attribute__((ext_vector_type(16))) __bf16 v16bf;
typedef __attribute__((ext_vector_type(8)))  float  v8f;
typedef int v4i_b128 __attribute__((vector_size(16)));

#define AS1 __attribute__((address_space(1)))
#define AS3 __attribute__((address_space(3)))

#if __has_builtin(__builtin_amdgcn_global_load_async_to_lds_b128)
#define HAVE_ASYNC_LDS 1
#else
#define HAVE_ASYNC_LDS 0
#endif

__device__ __forceinline__ void wait_async0() {
#if __has_builtin(__builtin_amdgcn_s_wait_asynccnt)
    __builtin_amdgcn_s_wait_asynccnt(0);
#else
    asm volatile("s_wait_asynccnt 0x0" ::: "memory");
#endif
}

#define BB     8192
#define STOCH  32
#define DISC   32
#define SS     1024
#define DETER  1024
#define ACTDIM 64
#define EMBED  4096
#define HID    1024

#define TMB 256          // block rows
#define TN  128          // block cols
#define TK  32           // K step

__device__ __forceinline__ unsigned short f32_to_bf16_bits(float f) {
    unsigned int u = __float_as_uint(f);
    unsigned int lsb = (u >> 16) & 1u;
    u += 0x7fffu + lsb;                 // round to nearest even
    return (unsigned short)(u >> 16);
}

__device__ __forceinline__ float sigmoidf_dev(float x) {
    return 1.0f / (1.0f + __expf(-x));
}

// ---------------------------------------------------------------------------
// Swizzled operand layouts (fragment order, per CDNA5 §7.12.2):
//  A[M,K] -> [mt=m/16][kt=k/32][q:2][lane:32][i8:8] halves:
//    k'=k%32: c=k'/8, h=c&1, q=c>>1, lane=h*16+m%16, i8=k'%8.
//  B[K,N] -> [nt=n/16][kt=k/32][q:2][lane:32][i8:8] halves:
//    k'=k%32: c=k'/8, half=c&1, q=c>>1, lane=half*16+n%16, i8=k'%8.
// One 16x32 (32x16) fragment = 1KB contiguous, 32 bytes per lane
// (quad q=0 at +0 halves, quad q=1 at +256 halves).
// ---------------------------------------------------------------------------
__device__ __forceinline__ size_t a_swz(int m, int k, int Kdim) {
    const int mt = m >> 4, kt = k >> 5, kp = k & 31;
    const int c = kp >> 3, h = c & 1, q = c >> 1, i8 = kp & 7;
    const int l = (h << 4) + (m & 15);
    return ((((size_t)mt * (Kdim >> 5) + kt) * 2 + q) * 32 + l) * 8 + i8;
}
__device__ __forceinline__ size_t b_swz(int k, int n, int Kdim) {
    const int nt = n >> 4, kt = k >> 5, kp = k & 31;
    const int c = kp >> 3, half = c & 1, q = c >> 1, i8 = kp & 7;
    const int l = (half << 4) + (n & 15);
    return ((((size_t)nt * (Kdim >> 5) + kt) * 2 + q) * 32 + l) * 8 + i8;
}

// ---------------------------------------------------------------------------
// Tiled bf16 WMMA GEMM on swizzled operands.
//   C = act(A * B + bias);  ACT: 0=none, 1=SiLU.
//   OBF16=1: store bf16 bits A-swizzled for the next GEMM (Kdim = Knext).
//   OBF16=0: store f32 row-major with leading dim ldo.
// Block = 256 threads = 8 waves; tile 256x128x32; wave w owns a 64x64
// sub-tile (wr=w>>1, wc=w&1) -> 4x4 WMMA ops per K-step.
// ---------------------------------------------------------------------------
template<int ACT, int OBF16>
__global__ __launch_bounds__(256, 1)
void gemm_bf16_wmma(const unsigned short* __restrict__ A,
                    const unsigned short* __restrict__ Bm,
                    const float* __restrict__ bias,
                    void* __restrict__ outp, int ldo, int Knext,
                    int M, int N, int K)
{
    // A: 16 tiles x 1KB, B: 8 tiles x 1KB; double buffered (48KB total).
    __shared__ __align__(16) unsigned short lA[2][(TMB / 16) * 512];
    __shared__ __align__(16) unsigned short lB[2][(TN / 16) * 512];

    const int tid  = threadIdx.x;
    const int wave = tid >> 5;
    const int lane = tid & 31;
    const int m0   = blockIdx.y * TMB;
    const int n0   = blockIdx.x * TN;
    const int wr   = wave >> 1;      // 0..3 : 64 rows each
    const int wc   = wave & 1;       // 0..1 : 64 cols each
    const int nkt  = K >> 5;

    // staging thread mapping
    const int taA = tid >> 4;        // A tile 0..15 (16 threads/tile)
    const int waA = tid & 15;        //   64B each, 4 x 16B quads
    const int taB = tid >> 5;        // B tile 0..7  (32 threads/tile)
    const int laB = tid & 31;        //   32B each, 2 x 16B quads

    v8f acc[4][4];
    #pragma unroll
    for (int i = 0; i < 4; i++)
        #pragma unroll
        for (int j = 0; j < 4; j++)
            #pragma unroll
            for (int r = 0; r < 8; r++) acc[i][j][r] = 0.0f;

#if HAVE_ASYNC_LDS
    // -------- async staging: global -> LDS with no VGPR transit ----------
    auto stage_async = [&](int kt, int buf) {
        const size_t abase = ((size_t)(m0 / 16 + taA) * nkt + kt) * 512;
        #pragma unroll
        for (int j = 0; j < 4; j++)
            __builtin_amdgcn_global_load_async_to_lds_b128(
                (AS1 v4i_b128*)(A + abase + (j * 16 + waA) * 8),
                (AS3 v4i_b128*)&lA[buf][taA * 512 + (j * 16 + waA) * 8], 0, 0);
        const size_t bbase = ((size_t)(n0 / 16 + taB) * nkt + kt) * 512;
        __builtin_amdgcn_global_load_async_to_lds_b128(
            (AS1 v4i_b128*)(Bm + bbase + laB * 8),
            (AS3 v4i_b128*)&lB[buf][taB * 512 + laB * 8], 0, 0);
        __builtin_amdgcn_global_load_async_to_lds_b128(
            (AS1 v4i_b128*)(Bm + bbase + 256 + laB * 8),
            (AS3 v4i_b128*)&lB[buf][taB * 512 + 256 + laB * 8], 0, 0);
        if (kt + 1 < nkt) {
            __builtin_prefetch(A + abase + 512, 0, 1);
            __builtin_prefetch(Bm + bbase + 512, 0, 1);
        }
    };
    stage_async(0, 0);
#else
    // -------- fallback: named-register pipeline (no arrays, no lambdas) --
    uint4 sa0, sa1, sa2, sa3, sb0, sb1;
    {
        const size_t abase = ((size_t)(m0 / 16 + taA) * nkt) * 512;
        sa0 = *(const uint4*)(A + abase + (0 * 16 + waA) * 8);
        sa1 = *(const uint4*)(A + abase + (1 * 16 + waA) * 8);
        sa2 = *(const uint4*)(A + abase + (2 * 16 + waA) * 8);
        sa3 = *(const uint4*)(A + abase + (3 * 16 + waA) * 8);
        const size_t bbase = ((size_t)(n0 / 16 + taB) * nkt) * 512;
        sb0 = *(const uint4*)(Bm + bbase + laB * 8);
        sb1 = *(const uint4*)(Bm + bbase + 256 + laB * 8);
        *(uint4*)&lA[0][taA * 512 + (0 * 16 + waA) * 8] = sa0;
        *(uint4*)&lA[0][taA * 512 + (1 * 16 + waA) * 8] = sa1;
        *(uint4*)&lA[0][taA * 512 + (2 * 16 + waA) * 8] = sa2;
        *(uint4*)&lA[0][taA * 512 + (3 * 16 + waA) * 8] = sa3;
        *(uint4*)&lB[0][taB * 512 + laB * 8]       = sb0;
        *(uint4*)&lB[0][taB * 512 + 256 + laB * 8] = sb1;
    }
#endif

    for (int kt = 0; kt < nkt; kt++) {
        const int buf = kt & 1;
#if HAVE_ASYNC_LDS
        wait_async0();                         // this wave's async LDS writes done
#endif
        __syncthreads();                       // staging of tile kt visible
        const bool more = (kt + 1 < nkt);
#if HAVE_ASYNC_LDS
        if (more) stage_async(kt + 1, buf ^ 1);
#else
        if (more) {
            const size_t abase = ((size_t)(m0 / 16 + taA) * nkt + kt + 1) * 512;
            sa0 = *(const uint4*)(A + abase + (0 * 16 + waA) * 8);
            sa1 = *(const uint4*)(A + abase + (1 * 16 + waA) * 8);
            sa2 = *(const uint4*)(A + abase + (2 * 16 + waA) * 8);
            sa3 = *(const uint4*)(A + abase + (3 * 16 + waA) * 8);
            const size_t bbase = ((size_t)(n0 / 16 + taB) * nkt + kt + 1) * 512;
            sb0 = *(const uint4*)(Bm + bbase + laB * 8);
            sb1 = *(const uint4*)(Bm + bbase + 256 + laB * 8);
            __builtin_prefetch(A + abase + 512, 0, 1);
            __builtin_prefetch(Bm + bbase + 512, 0, 1);
        }
#endif

        // ---- MAC block: 2 m-groups of 2 fragments (caps VGPR pressure) ----
        #pragma unroll
        for (int g = 0; g < 2; g++) {
            const int mt = wr * 4 + g * 2;
            union { v16bf v; uint4 q[2]; } fa0, fa1;
            fa0.q[0] = *(const uint4*)&lA[buf][mt * 512 + lane * 8];
            fa0.q[1] = *(const uint4*)&lA[buf][mt * 512 + 256 + lane * 8];
            fa1.q[0] = *(const uint4*)&lA[buf][(mt + 1) * 512 + lane * 8];
            fa1.q[1] = *(const uint4*)&lA[buf][(mt + 1) * 512 + 256 + lane * 8];
            #pragma unroll
            for (int ni = 0; ni < 4; ni++) {
                const int nt = wc * 4 + ni;
                union { v16bf v; uint4 q[2]; } fb;
                fb.q[0] = *(const uint4*)&lB[buf][nt * 512 + lane * 8];
                fb.q[1] = *(const uint4*)&lB[buf][nt * 512 + 256 + lane * 8];
                acc[g * 2][ni] = __builtin_amdgcn_wmma_f32_16x16x32_bf16(
                    false, fa0.v, false, fb.v,
                    (short)0, acc[g * 2][ni], false, false);
                acc[g * 2 + 1][ni] = __builtin_amdgcn_wmma_f32_16x16x32_bf16(
                    false, fa1.v, false, fb.v,
                    (short)0, acc[g * 2 + 1][ni], false, false);
            }
        }

#if !HAVE_ASYNC_LDS
        if (more) {
            const int nb = buf ^ 1;
            *(uint4*)&lA[nb][taA * 512 + (0 * 16 + waA) * 8] = sa0;
            *(uint4*)&lA[nb][taA * 512 + (1 * 16 + waA) * 8] = sa1;
            *(uint4*)&lA[nb][taA * 512 + (2 * 16 + waA) * 8] = sa2;
            *(uint4*)&lA[nb][taA * 512 + (3 * 16 + waA) * 8] = sa3;
            *(uint4*)&lB[nb][taB * 512 + laB * 8]       = sb0;
            *(uint4*)&lB[nb][taB * 512 + 256 + laB * 8] = sb1;
        }
#endif
    }

    // ---- epilogue: C/D layout VGPR r -> row r + 8*half, col = lane%16 ----
    const int halfl = lane >> 4;
    const int lmod  = lane & 15;
    #pragma unroll
    for (int mi = 0; mi < 4; mi++) {
        #pragma unroll
        for (int ni = 0; ni < 4; ni++) {
            const int col = n0 + wc * 64 + ni * 16 + lmod;
            const float bv = bias[col];
            #pragma unroll
            for (int r = 0; r < 8; r++) {
                const int rowg = m0 + wr * 64 + mi * 16 + r + 8 * halfl;
                float v = acc[mi][ni][r] + bv;
                if (ACT == 1) v = v * sigmoidf_dev(v);
                if (OBF16)
                    ((unsigned short*)outp)[a_swz(rowg, col, Knext)] = f32_to_bf16_bits(v);
                else
                    ((float*)outp)[(size_t)rowg * ldo + col] = v;
            }
        }
    }
}

// ---------------------------------------------------------------------------
// Prep kernels (all emit swizzled bf16)
// ---------------------------------------------------------------------------
__global__ void cvt_weight_swz(const float* __restrict__ in,
                               unsigned short* __restrict__ out, int K, int N)
{
    size_t i = (size_t)blockIdx.x * 256 + threadIdx.x;
    if (i >= (size_t)K * N) return;
    const int k = (int)(i / N), n = (int)(i % N);
    out[b_swz(k, n, K)] = f32_to_bf16_bits(in[i]);
}

// A0[B,1088] = keep * [posterior | action]   (A-swizzled, Kdim=1088)
__global__ void build_A0(const float* __restrict__ post, const float* __restrict__ act,
                         const float* __restrict__ is_first, unsigned short* __restrict__ A0)
{
    size_t i = (size_t)blockIdx.x * 256 + threadIdx.x;
    if (i >= (size_t)BB * (SS + ACTDIM)) return;
    const int b = (int)(i / (SS + ACTDIM));
    const int c = (int)(i % (SS + ACTDIM));
    const float keep = 1.0f - is_first[b];
    const float v = (c < SS) ? post[(size_t)b * SS + c]
                             : act[(size_t)b * ACTDIM + (c - SS)];
    A0[a_swz(b, c, SS + ACTDIM)] = f32_to_bf16_bits(keep * v);
}

// A1[b, 1024+j] = bf16(keep * h[b,j])   (A-swizzled, Kdim=2048)
__global__ void build_A1_h(const float* __restrict__ h, const float* __restrict__ is_first,
                           unsigned short* __restrict__ A1)
{
    size_t i = (size_t)blockIdx.x * 256 + threadIdx.x;
    if (i >= (size_t)BB * DETER) return;
    const int b = (int)(i / DETER);
    const int j = (int)(i % DETER);
    const float keep = 1.0f - is_first[b];
    A1[a_swz(b, DETER + j, HID + DETER)] = f32_to_bf16_bits(keep * h[i]);
}

// Ar[b, 1024+j] = bf16(embed[b,j])      (A-swizzled, Kdim=5120)
__global__ void build_Ar_embed(const float* __restrict__ emb, unsigned short* __restrict__ Ar)
{
    size_t i = (size_t)blockIdx.x * 256 + threadIdx.x;
    if (i >= (size_t)BB * EMBED) return;
    const int b = (int)(i / EMBED);
    const int j = (int)(i % EMBED);
    Ar[a_swz(b, DETER + j, DETER + EMBED)] = f32_to_bf16_bits(emb[i]);
}

// ---------------------------------------------------------------------------
// LayerNorm-GRU pointwise: one block per batch row.
// ---------------------------------------------------------------------------
__global__ __launch_bounds__(256)
void gru_pointwise(const float* __restrict__ parts,
                   const float* __restrict__ ln_g, const float* __restrict__ ln_b,
                   const float* __restrict__ recurrent, const float* __restrict__ is_first,
                   float* __restrict__ hnew_out,
                   unsigned short* __restrict__ At,   // A-swizzled, Kdim=1024
                   unsigned short* __restrict__ Ar)   // A-swizzled, Kdim=5120, cols 0..1023
{
    const int b   = blockIdx.x;
    const int tid = threadIdx.x;
    const float* p = parts + (size_t)b * (3 * DETER);
    __shared__ float red[256];

    float s = 0.0f;
    for (int j = tid; j < 3 * DETER; j += 256) s += p[j];
    red[tid] = s; __syncthreads();
    for (int off = 128; off > 0; off >>= 1) {
        if (tid < off) red[tid] += red[tid + off];
        __syncthreads();
    }
    const float mean = red[0] * (1.0f / (3 * DETER));
    __syncthreads();

    float v = 0.0f;
    for (int j = tid; j < 3 * DETER; j += 256) { float d = p[j] - mean; v += d * d; }
    red[tid] = v; __syncthreads();
    for (int off = 128; off > 0; off >>= 1) {
        if (tid < off) red[tid] += red[tid + off];
        __syncthreads();
    }
    const float rstd = rsqrtf(red[0] * (1.0f / (3 * DETER)) + 1e-5f);
    const float keep = 1.0f - is_first[b];

    for (int j = tid; j < DETER; j += 256) {
        float r = ln_g[j]             * (p[j]             - mean) * rstd + ln_b[j];
        float c = ln_g[j + DETER]     * (p[j + DETER]     - mean) * rstd + ln_b[j + DETER];
        float u = ln_g[j + 2 * DETER] * (p[j + 2 * DETER] - mean) * rstd + ln_b[j + 2 * DETER];
        r = sigmoidf_dev(r);
        c = tanhf(r * c);
        u = sigmoidf_dev(u - 1.0f);
        const float h  = keep * recurrent[(size_t)b * DETER + j];
        const float hn = u * c + (1.0f - u) * h;
        hnew_out[(size_t)b * DETER + j] = hn;
        const unsigned short hb = f32_to_bf16_bits(hn);
        At[a_swz(b, j, DETER)]         = hb;
        Ar[a_swz(b, j, DETER + EMBED)] = hb;
    }
}

// ---------------------------------------------------------------------------
// Straight-through categorical: hard + probs - sg(probs) == one_hot(sample).
// ---------------------------------------------------------------------------
__device__ __forceinline__ float hash_uniform(unsigned int x) {
    x ^= x >> 17; x *= 0xed5ad4bbu;
    x ^= x >> 11; x *= 0xac4c1b51u;
    x ^= x >> 15; x *= 0x31848babu;
    x ^= x >> 14;
    return ((x >> 8) + 0.5f) * (1.0f / 16777216.0f);
}

__global__ void sample_onehot(const float* __restrict__ logits, float* __restrict__ out)
{
    const int g = blockIdx.x * 256 + threadIdx.x;     // b*STOCH + var
    if (g >= BB * STOCH) return;
    const float* l = logits + (size_t)g * DISC;
    float mx = -1e30f;
    #pragma unroll
    for (int i = 0; i < DISC; i++) mx = fmaxf(mx, l[i]);
    float e[DISC]; float sum = 0.0f;
    #pragma unroll
    for (int i = 0; i < DISC; i++) { e[i] = __expf(l[i] - mx); sum += e[i]; }
    const float u = hash_uniform((unsigned int)g * 2654435761u + 42u) * sum;
    int idx = DISC - 1; float c = 0.0f;
    #pragma unroll
    for (int i = 0; i < DISC; i++) { c += e[i]; if (u < c) { idx = i; break; } }
    float* o = out + (size_t)g * DISC;
    #pragma unroll
    for (int i = 0; i < DISC; i++) o[i] = (i == idx) ? 1.0f : 0.0f;
}

// ---------------------------------------------------------------------------
// Host-side pipeline
// ---------------------------------------------------------------------------
extern "C" void kernel_launch(void* const* d_in, const int* in_sizes, int n_in,
                              void* d_out, int out_size, void* d_ws, size_t ws_size,
                              hipStream_t stream)
{
    const float* posterior = (const float*)d_in[0];
    const float* recurrent = (const float*)d_in[1];
    const float* action    = (const float*)d_in[2];
    const float* embedded  = (const float*)d_in[3];
    const float* is_first  = (const float*)d_in[4];
    const float* W_pre = (const float*)d_in[5];  const float* b_pre = (const float*)d_in[6];
    const float* W_gru = (const float*)d_in[7];  const float* b_gru = (const float*)d_in[8];
    const float* ln_g  = (const float*)d_in[9];  const float* ln_b  = (const float*)d_in[10];
    const float* W_r1  = (const float*)d_in[11]; const float* b_r1  = (const float*)d_in[12];
    const float* W_r2  = (const float*)d_in[13]; const float* b_r2  = (const float*)d_in[14];
    const float* W_t1  = (const float*)d_in[15]; const float* b_t1  = (const float*)d_in[16];
    const float* W_t2  = (const float*)d_in[17]; const float* b_t2  = (const float*)d_in[18];

    float* out = (float*)d_out;
    float* out_hnew  = out;                         // [B,1024]
    float* out_post  = out + (size_t)BB * SS;       // [B,32,32]
    float* out_plog  = out + 2 * (size_t)BB * SS;   // [B,1024]
    float* out_tlog  = out + 3 * (size_t)BB * SS;   // [B,1024]

    // ---- carve workspace (256B aligned regions) ----
    size_t off = 0;
    auto carve = [&](size_t bytes) -> void* {
        void* p = (char*)d_ws + off;
        off += (bytes + 255) & ~(size_t)255;
        return p;
    };
    const int KPRE = SS + ACTDIM;          // 1088
    const int KGRU = HID + DETER;          // 2048
    const int KREP = DETER + EMBED;        // 5120

    unsigned short* Wpre_b = (unsigned short*)carve((size_t)KPRE * HID * 2);
    unsigned short* Wgru_b = (unsigned short*)carve((size_t)KGRU * 3 * DETER * 2);
    unsigned short* Wr1_b  = (unsigned short*)carve((size_t)KREP * HID * 2);
    unsigned short* Wr2_b  = (unsigned short*)carve((size_t)HID * SS * 2);
    unsigned short* Wt1_b  = (unsigned short*)carve((size_t)DETER * HID * 2);
    unsigned short* Wt2_b  = (unsigned short*)carve((size_t)HID * SS * 2);
    unsigned short* A0     = (unsigned short*)carve((size_t)BB * KPRE * 2);
    unsigned short* A1     = (unsigned short*)carve((size_t)BB * KGRU * 2);
    unsigned short* Ar     = (unsigned short*)carve((size_t)BB * KREP * 2);
    unsigned short* At     = (unsigned short*)carve((size_t)BB * DETER * 2);
    unsigned short* thid   = (unsigned short*)carve((size_t)BB * HID * 2);
    unsigned short* rhid   = (unsigned short*)carve((size_t)BB * HID * 2);
    float*          parts  = (float*)carve((size_t)BB * 3 * DETER * 4);
    (void)ws_size; (void)in_sizes; (void)n_in; (void)out_size;

    auto nblk = [](size_t n) { return (unsigned)((n + 255) / 256); };

    // ---- weight conversions f32 -> bf16 (B-swizzled) ----
    cvt_weight_swz<<<nblk((size_t)KPRE * HID), 256, 0, stream>>>(W_pre, Wpre_b, KPRE, HID);
    cvt_weight_swz<<<nblk((size_t)KGRU * 3 * DETER), 256, 0, stream>>>(W_gru, Wgru_b, KGRU, 3 * DETER);
    cvt_weight_swz<<<nblk((size_t)KREP * HID), 256, 0, stream>>>(W_r1, Wr1_b, KREP, HID);
    cvt_weight_swz<<<nblk((size_t)HID * SS), 256, 0, stream>>>(W_r2, Wr2_b, HID, SS);
    cvt_weight_swz<<<nblk((size_t)DETER * HID), 256, 0, stream>>>(W_t1, Wt1_b, DETER, HID);
    cvt_weight_swz<<<nblk((size_t)HID * SS), 256, 0, stream>>>(W_t2, Wt2_b, HID, SS);

    // ---- activation staging (A-swizzled) ----
    build_A0<<<nblk((size_t)BB * KPRE), 256, 0, stream>>>(posterior, action, is_first, A0);
    build_A1_h<<<nblk((size_t)BB * DETER), 256, 0, stream>>>(recurrent, is_first, A1);
    build_Ar_embed<<<nblk((size_t)BB * EMBED), 256, 0, stream>>>(embedded, Ar);

    // ---- GEMM 1: pre-MLP  (SiLU, bf16 -> A1 cols 0..1023, Knext=2048) ----
    gemm_bf16_wmma<1, 1><<<dim3(HID / TN, BB / TMB), 256, 0, stream>>>(
        A0, Wpre_b, b_pre, A1, 0, KGRU, BB, HID, KPRE);

    // ---- GEMM 2: GRU matmul (f32 out, no act) ----
    gemm_bf16_wmma<0, 0><<<dim3(3 * DETER / TN, BB / TMB), 256, 0, stream>>>(
        A1, Wgru_b, b_gru, parts, 3 * DETER, 0, BB, 3 * DETER, KGRU);

    // ---- LayerNorm + GRU gates -> h_new ----
    gru_pointwise<<<BB, 256, 0, stream>>>(parts, ln_g, ln_b, recurrent, is_first,
                                          out_hnew, At, Ar);

    // ---- transition head ----
    gemm_bf16_wmma<1, 1><<<dim3(HID / TN, BB / TMB), 256, 0, stream>>>(
        At, Wt1_b, b_t1, thid, 0, HID, BB, HID, DETER);
    gemm_bf16_wmma<0, 0><<<dim3(SS / TN, BB / TMB), 256, 0, stream>>>(
        thid, Wt2_b, b_t2, out_tlog, SS, 0, BB, SS, HID);

    // ---- representation head ----
    gemm_bf16_wmma<1, 1><<<dim3(HID / TN, BB / TMB), 256, 0, stream>>>(
        Ar, Wr1_b, b_r1, rhid, 0, HID, BB, HID, KREP);
    gemm_bf16_wmma<0, 0><<<dim3(SS / TN, BB / TMB), 256, 0, stream>>>(
        rhid, Wr2_b, b_r2, out_plog, SS, 0, BB, SS, HID);

    // ---- categorical straight-through sample (== one-hot) ----
    sample_onehot<<<nblk((size_t)BB * STOCH), 256, 0, stream>>>(out_plog, out_post);
}
